// DGNLayer_47425028882653
// MI455X (gfx1250) — matrix operand order repaired and checked by reading, not compiled, not run on hardware
//
#include <hip/hip_runtime.h>
#include <hip/hip_bf16.h>

// ---------------------------------------------------------------------------
// DGN layer for MI455X (gfx1250, wave32, WMMA).
// Fused: edge-GEMM (bf16 WMMA, f32 acc) + 3 segment scatters; node-GEMM
// (bf16 WMMA) + BN stats; BN/ReLU/residual epilogue. 'e' (819MB) is never
// materialized. Packed W1 (256KB bf16) is staged once per persistent block
// in LDS (320KB/WGP) so the edge GEMM's B-fragments come from ds_load,
// keeping L2 bandwidth for the h-gathers and the scatter atomics.
// ---------------------------------------------------------------------------

#define N_NODES  50000
#define N_EDGES  800000
#define DFEAT    256
#define EPS_W    1e-8f
#define BN_EPS   1e-5f

#define W1P_USHORTS (16 * 16 * 32 * 16)           // 131072 bf16 = 256 KB
#define EDGE_GRID   512                           // persistent blocks

typedef __attribute__((ext_vector_type(16))) __bf16 v16bf;
typedef __attribute__((ext_vector_type(8)))  float  v8f;

__device__ __forceinline__ unsigned short f2bf_u(float x) {
    union { float f; unsigned u; } v; v.f = x;
    unsigned r = v.u + 0x7FFFu + ((v.u >> 16) & 1u);   // round-to-nearest-even
    return (unsigned short)(r >> 16);
}

// ---------------------------------------------------------------------------
__global__ void zero_ws_kernel(float* __restrict__ p, long long n) {
    long long i = (long long)blockIdx.x * blockDim.x + threadIdx.x;
    long long stride = (long long)gridDim.x * blockDim.x;
    for (; i < n; i += stride) p[i] = 0.0f;
}

__global__ void cvt_h_bf16_kernel(const float* __restrict__ h,
                                  unsigned short* __restrict__ hbf, int n) {
    int i = blockIdx.x * blockDim.x + threadIdx.x;
    if (i < n) hbf[i] = f2bf_u(h[i]);
}

// Pack row-major W (KT*32 x 256, f32) into wave32 WMMA B-fragment layout:
// packed[((kt*16+nt)*32 + lane)*16 + j] = bf16( W[(kt*32 + (lane>=16?8:0) + (j<8?j:j+8))*256 + nt*16 + (lane&15)] )
__global__ void pack_w_kernel(const float* __restrict__ W,
                              unsigned short* __restrict__ wp, int KT) {
    int i = blockIdx.x * blockDim.x + threadIdx.x;
    int total = KT * 16 * 32 * 16;
    if (i >= total) return;
    int j    = i & 15;
    int lane = (i >> 4) & 31;
    int nt   = (i >> 9) & 15;
    int kt   = i >> 13;
    int k = kt * 32 + ((lane >= 16) ? 8 : 0) + ((j < 8) ? j : j + 8);
    int n = nt * 16 + (lane & 15);
    wp[i] = f2bf_u(W[k * DFEAT + n]);
}

// deg[dst] += 1 ; sabs[dst] += |eig[e,0]|
__global__ void edge_stats_kernel(const int* __restrict__ dst,
                                  const float* __restrict__ eig,
                                  float* __restrict__ deg,
                                  float* __restrict__ sabs, int n) {
    int i = blockIdx.x * blockDim.x + threadIdx.x;
    if (i >= n) return;
    int d = dst[i];
    atomicAdd(&deg[d], 1.0f);
    atomicAdd(&sabs[d], fabsf(eig[2 * i]));
}

// in-place: deg -> 1/max(deg,1) ; sabs -> 1/(sabs+EPS)
__global__ void node_finalize_kernel(float* __restrict__ deg,
                                     float* __restrict__ sabs, int n) {
    int i = blockIdx.x * blockDim.x + threadIdx.x;
    if (i >= n) return;
    deg[i]  = 1.0f / fmaxf(deg[i], 1.0f);
    sabs[i] = 1.0f / (sabs[i] + EPS_W);
}

// ---------------------------------------------------------------------------
// Edge GEMM + scatter (persistent blocks, W1 B-fragments staged in LDS).
// Per wave, one tile of 16 edges x 256 outputs per loop iteration.
// e = [h[src] | h[dst]] @ W1 + b1 via 16x16x32 bf16 WMMA (K=512), then
// scattered: accm += e, accav += e*|wn|, accdx += e*wn  (to dst node).
// ---------------------------------------------------------------------------
__global__ __launch_bounds__(256) void edge_gemm_scatter_kernel(
    const unsigned short* __restrict__ hbf,   // N_NODES x 256 bf16
    const unsigned short* __restrict__ w1p,   // packed B frags, 16kt x 16nt
    const float* __restrict__ b1,
    const float* __restrict__ eig,
    const int*   __restrict__ src,
    const int*   __restrict__ dst,
    const float* __restrict__ sinv,           // 1/(seg|w|+eps) per node
    float* __restrict__ accm,
    float* __restrict__ accav,
    float* __restrict__ accdx)
{
    extern __shared__ unsigned short w1s[];   // 256 KB: full packed W1

    // --- one-time cooperative fill of LDS with packed W1 (uint4 granules) ---
    {
        const uint4* gsrc = (const uint4*)w1p;
        uint4* ldst = (uint4*)w1s;
        const int n4 = W1P_USHORTS / 8;       // 16384 uint4
        for (int i = threadIdx.x; i < n4; i += 256) ldst[i] = gsrc[i];
    }
    __syncthreads();

    const int lane = threadIdx.x & 31;
    const int wave = threadIdx.x >> 5;

    const int arow = lane & 15;     // A-fragment row (edge within tile)
    const int hi   = lane >> 4;     // 0: K-group low, 1: K-group high
    const int koff = hi * 8;
    const int rbase = hi * 8;       // C/D rows owned by this lane

    const int ntiles = N_EDGES / 16;
    for (int tile = blockIdx.x * 8 + wave; tile < ntiles;
         tile += EDGE_GRID * 8) {
        const int eb = tile * 16;

        const int s_node = src[eb + arow];
        const int d_node = dst[eb + arow];

        v8f zero = {};
        v8f acc[16];
#pragma unroll
        for (int nt = 0; nt < 16; ++nt) acc[nt] = zero;

        for (int kt = 0; kt < 16; ++kt) {
            // K-block of 32; kt<8 -> h[src] half, else h[dst] half
            const int node = (kt < 8) ? s_node : d_node;
            const int kk   = (kt & 7) * 32 + koff;
            const unsigned short* ap = hbf + (long long)node * DFEAT + kk;
            union { uint4 q[2]; v16bf v; } au;
            au.q[0] = *(const uint4*)(ap);        // K = kk .. kk+7
            au.q[1] = *(const uint4*)(ap + 16);   // K = kk+16 .. kk+23
#pragma unroll
            for (int nt = 0; nt < 16; ++nt) {
                const unsigned short* bp = w1s + (((kt * 16 + nt) * 32 + lane) << 4);
                union { uint4 q[2]; v16bf v; } bu;
                bu.q[0] = *(const uint4*)(bp);        // ds_load_b128
                bu.q[1] = *(const uint4*)(bp + 8);
                acc[nt] = __builtin_amdgcn_wmma_f32_16x16x32_bf16(
                    false, au.v, false, bu.v, (short)0, acc[nt], false, false);
            }
        }

        // C/D layout: VGPR r of lane L -> row M = r + 8*(L>=16), col N = L&15.
        int   dnode[8];
        float wnv[8], awnv[8];
#pragma unroll
        for (int r = 0; r < 8; ++r) {
            const int m  = rbase + r;
            const int dn = dst[eb + m];
            dnode[r] = dn;
            const float wn = eig[2 * (eb + m)] * sinv[dn];
            wnv[r]  = wn;
            awnv[r] = fabsf(wn);
        }

#pragma unroll
        for (int nt = 0; nt < 16; ++nt) {
            const int   n   = nt * 16 + (lane & 15);
            const float b1n = b1[n];
#pragma unroll
            for (int r = 0; r < 8; ++r) {
                const float val = acc[nt][r] + b1n;
                const long long o = (long long)dnode[r] * DFEAT + n;
                atomicAdd(accm  + o, val);
                atomicAdd(accav + o, val * awnv[r]);
                atomicAdd(accdx + o, val * wnv[r]);
            }
        }
    }
}

// ---------------------------------------------------------------------------
// Node GEMM + BN stats: out_pre = hc @ W2 + b2 scaled by snorm, where
// hc = [h, accm*rdeg, accav, |accdx - h|] (K=1024), built on the fly in bf16.
// Also accumulates per-column sum / sum-of-squares for batch-norm.
// W2 packed (512KB) exceeds LDS, and this GEMM's B traffic (~1.6GB) is
// modest, so B-fragments stream from L2.
// ---------------------------------------------------------------------------
__global__ __launch_bounds__(256) void node_gemm_bnstats_kernel(
    const float* __restrict__ h,
    const unsigned short* __restrict__ w2p,   // packed B frags, 32kt x 16nt
    const float* __restrict__ b2,
    const float* __restrict__ rdeg,
    const float* __restrict__ accm,
    const float* __restrict__ accav,
    const float* __restrict__ accdx,
    const float* __restrict__ snorm,
    float* __restrict__ outpre,
    float* __restrict__ colsum,
    float* __restrict__ colsq)
{
    const int lane = threadIdx.x & 31;
    const int wave = threadIdx.x >> 5;
    const int tile = blockIdx.x * 8 + wave;
    if (tile >= N_NODES / 16) return;
    const int nb = tile * 16;

    const int arow  = lane & 15;
    const int hi    = lane >> 4;
    const int koff  = hi * 8;
    const int anode = nb + arow;
    const long long rowbase = (long long)anode * DFEAT;
    const float rdg = rdeg[anode];

    v8f zero = {};
    v8f acc[16];
#pragma unroll
    for (int nt = 0; nt < 16; ++nt) acc[nt] = zero;

    for (int kt = 0; kt < 32; ++kt) {
        const int region = kt >> 3;                 // 0:h 1:mean 2:av 3:dx
        const int kbase  = (kt & 7) * 32 + koff;
        union { unsigned u[8]; v16bf v; } au;
#pragma unroll
        for (int d8 = 0; d8 < 8; ++d8) {
            const int ja = 2 * d8, jb = 2 * d8 + 1;
            const int ka = kbase + ((ja < 8) ? ja : ja + 8);
            const int kb = kbase + ((jb < 8) ? jb : jb + 8);
            const long long ia = rowbase + ka, ib = rowbase + kb;
            float xa, xb;
            if (region == 0)      { xa = h[ia];                      xb = h[ib]; }
            else if (region == 1) { xa = accm[ia] * rdg;             xb = accm[ib] * rdg; }
            else if (region == 2) { xa = accav[ia];                  xb = accav[ib]; }
            else                  { xa = fabsf(accdx[ia] - h[ia]);   xb = fabsf(accdx[ib] - h[ib]); }
            au.u[d8] = (unsigned)f2bf_u(xa) | ((unsigned)f2bf_u(xb) << 16);
        }
#pragma unroll
        for (int nt = 0; nt < 16; ++nt) {
            const unsigned short* bp = w2p + (((kt * 16 + nt) * 32 + lane) << 4);
            union { uint4 q[2]; v16bf v; } bu;
            bu.q[0] = *(const uint4*)(bp);
            bu.q[1] = *(const uint4*)(bp + 8);
            acc[nt] = __builtin_amdgcn_wmma_f32_16x16x32_bf16(
                false, au.v, false, bu.v, (short)0, acc[nt], false, false);
        }
    }

    float snm[8];
#pragma unroll
    for (int r = 0; r < 8; ++r) snm[r] = snorm[nb + hi * 8 + r];

#pragma unroll
    for (int nt = 0; nt < 16; ++nt) {
        const int   n   = nt * 16 + (lane & 15);
        const float b2n = b2[n];
        float ls = 0.0f, lq = 0.0f;
#pragma unroll
        for (int r = 0; r < 8; ++r) {
            const int node = nb + hi * 8 + r;
            const float val = (acc[nt][r] + b2n) * snm[r];
            outpre[(long long)node * DFEAT + n] = val;
            ls += val;
            lq += val * val;
        }
        atomicAdd(&colsum[n], ls);
        atomicAdd(&colsq[n], lq);
    }
}

// ---------------------------------------------------------------------------
__global__ void bn_relu_residual_kernel(
    const float* __restrict__ outpre, const float* __restrict__ h,
    const float* __restrict__ colsum, const float* __restrict__ colsq,
    const float* __restrict__ gamma,  const float* __restrict__ beta,
    float* __restrict__ out)
{
    long long i = (long long)blockIdx.x * blockDim.x + threadIdx.x;
    if (i >= (long long)N_NODES * DFEAT) return;
    const int c = (int)(i & (DFEAT - 1));
    const float inv_n = 1.0f / (float)N_NODES;
    const float mu  = colsum[c] * inv_n;
    const float var = colsq[c] * inv_n - mu * mu;
    const float x = outpre[i];
    float y = (x - mu) * rsqrtf(var + BN_EPS) * gamma[c] + beta[c];
    y = fmaxf(y, 0.0f);
    out[i] = h[i] + y;
}

// ---------------------------------------------------------------------------
extern "C" void kernel_launch(void* const* d_in, const int* in_sizes, int n_in,
                              void* d_out, int out_size, void* d_ws, size_t ws_size,
                              hipStream_t stream) {
    (void)in_sizes; (void)n_in; (void)out_size; (void)ws_size;

    const float* h     = (const float*)d_in[0];
    const float* eig   = (const float*)d_in[1];
    const float* snorm = (const float*)d_in[2];
    const float* W1    = (const float*)d_in[3];
    const float* b1    = (const float*)d_in[4];
    const float* W2    = (const float*)d_in[5];
    const float* b2    = (const float*)d_in[6];
    const float* gamma = (const float*)d_in[7];
    const float* beta  = (const float*)d_in[8];
    const int*   src   = (const int*)d_in[9];
    const int*   dst   = (const int*)d_in[10];
    float* out = (float*)d_out;

    // Workspace layout (zeroed regions first, contiguous):
    const long long ND = (long long)N_NODES * DFEAT;        // 12,800,000
    float* deg    = (float*)d_ws;            // N_NODES
    float* sabs   = deg    + N_NODES;        // N_NODES
    float* accm   = sabs   + N_NODES;        // ND
    float* accav  = accm   + ND;             // ND
    float* accdx  = accav  + ND;             // ND
    float* colsum = accdx  + ND;             // 256
    float* colsq  = colsum + DFEAT;          // 256
    float* outpre = colsq  + DFEAT;          // ND  (fully overwritten)
    unsigned short* hbf = (unsigned short*)(outpre + ND);   // ND bf16
    unsigned short* w1p = hbf + ND;          // 512*256 bf16
    unsigned short* w2p = w1p + 512 * DFEAT; // 1024*256 bf16

    const long long nzero = 2LL * N_NODES + 3LL * ND + 2LL * DFEAT;

    // 1) zero all accumulator regions (atomics accumulate; must reset per call)
    zero_ws_kernel<<<2048, 256, 0, stream>>>((float*)d_ws, nzero);

    // 2) h -> bf16 (halves the L2 gather traffic of the edge GEMM)
    cvt_h_bf16_kernel<<<(int)((ND + 255) / 256), 256, 0, stream>>>(h, hbf, (int)ND);

    // 3) pack W1 (16 K-tiles) and W2 (32 K-tiles) into WMMA B-fragment layout
    pack_w_kernel<<<(16 * 16 * 32 * 16 + 255) / 256, 256, 0, stream>>>(W1, w1p, 16);
    pack_w_kernel<<<(32 * 16 * 32 * 16 + 255) / 256, 256, 0, stream>>>(W2, w2p, 32);

    // 4) per-node degree and sum|w| via atomics, then reciprocals
    edge_stats_kernel<<<(N_EDGES + 255) / 256, 256, 0, stream>>>(dst, eig, deg, sabs, N_EDGES);
    node_finalize_kernel<<<(N_NODES + 255) / 256, 256, 0, stream>>>(deg, sabs, N_NODES);

    // 5) fused edge GEMM + weighted segment scatters
    //    persistent blocks, full packed W1 (256KB) in dynamic LDS
    edge_gemm_scatter_kernel<<<EDGE_GRID, 256, W1P_USHORTS * sizeof(unsigned short), stream>>>(
        hbf, w1p, b1, eig, src, dst, sabs, accm, accav, accdx);

    // 6) fused node GEMM + snorm scale + BN statistics (3125 tiles)
    node_gemm_bnstats_kernel<<<(N_NODES / 16 + 7) / 8, 256, 0, stream>>>(
        h, w2p, b2, deg, accm, accav, accdx, snorm, outpre, colsum, colsq);

    // 7) batch-norm + ReLU + residual
    bn_relu_residual_kernel<<<(int)((ND + 255) / 256), 256, 0, stream>>>(
        outpre, h, colsum, colsq, gamma, beta, out);
}
